// GraphAttentionLayer_16527034155595
// MI455X (gfx1250) — matrix operand order repaired and checked by reading, not compiled, not run on hardware
//
#include <hip/hip_runtime.h>

typedef __attribute__((ext_vector_type(16))) _Float16 v16h;
typedef __attribute__((ext_vector_type(8)))  float    v8f;
typedef __attribute__((ext_vector_type(2)))  float    v2f;

#define NB   8
#define NN   1024
#define IND  128
#define OUTD 64
#define NH   4

// ---------------------------------------------------------------------------
// Kernel 1: Ht = H @ W  (M=8192, K=128, Nout=256), exact f32 WMMA 16x16x4.
// Writes Ht transposed as f16: HtT[(bh*64+d)*1024 + n].
// One wave per 16x16 output tile; 32 k-steps of V_WMMA_F32_16X16X4_F32.
// ---------------------------------------------------------------------------
__global__ __launch_bounds__(128) void k_proj(const float* __restrict__ Hm,
                                              const float* __restrict__ Wm,
                                              _Float16* __restrict__ HtT) {
  const int lane = threadIdx.x & 31, wave = threadIdx.x >> 5;
  const int tileId = blockIdx.x * 4 + wave;       // 8192 tiles total
  const int tile_n = tileId & 15;                 // 256/16 column tiles
  const int tile_m = tileId >> 4;                 // 8192/16 row tiles
  const int row  = lane & 15;
  const int half = lane >> 4;

  const int gm   = tile_m * 16 + row;             // A-operand row for this lane
  const int nout = tile_n * 16 + row;             // B/C column for this lane
  const float* __restrict__ hrow = Hm + (size_t)gm * IND;

  v8f acc = {};
#pragma unroll
  for (int kk = 0; kk < 32; ++kk) {
    const int k0 = kk * 4 + half * 2;
    // A 16x4 f32: VGPR0 = K0|K2, VGPR1 = K1|K3 (lane = M)
    v2f a; a[0] = hrow[k0]; a[1] = hrow[k0 + 1];
    // B 4x16 f32: VGPR0 = K0|K2, VGPR1 = K1|K3 (lane = N)
    v2f b; b[0] = Wm[(size_t)k0 * 256 + nout];
           b[1] = Wm[(size_t)(k0 + 1) * 256 + nout];
    acc = __builtin_amdgcn_wmma_f32_16x16x4_f32(false, a, false, b,
                                                (short)0, acc, false, false);
  }

  const int h = nout >> 6, d = nout & 63;
#pragma unroll
  for (int i = 0; i < 8; ++i) {
    const int gmo = tile_m * 16 + i + half * 8;   // C/D row for VGPR i
    const int b  = gmo >> 10, n = gmo & 1023;
    const int bh = b * NH + h;
    HtT[(size_t)(bh * 64 + d) * 1024 + n] = (_Float16)acc[i];
  }
}

// ---------------------------------------------------------------------------
// Kernel 1b: e_src/e_dst = Ht . a_src/a_dst, plus per-(b,h) max of e_dst
// (upper bound used for the single-pass stable softmax).
// ---------------------------------------------------------------------------
__global__ __launch_bounds__(256) void k_edge(const _Float16* __restrict__ HtT,
                                              const float* __restrict__ av,
                                              float* __restrict__ esrc,
                                              float* __restrict__ edst,
                                              float* __restrict__ maxd) {
  __shared__ float as_[64], ad_[64], red[256];
  const int bh = blockIdx.x, h = bh & 3, t = threadIdx.x;
  if (t < 64)        as_[t]      = av[h * 128 + t];
  else if (t < 128)  ad_[t - 64] = av[h * 128 + 64 + (t - 64)];
  __syncthreads();

  float tmax = -1e30f;
  for (int n = t; n < 1024; n += 256) {
    float s = 0.f, dd = 0.f;
    const _Float16* __restrict__ base = HtT + (size_t)bh * 64 * 1024 + n;
#pragma unroll 8
    for (int d = 0; d < 64; ++d) {
      const float v = (float)base[(size_t)d * 1024];
      s  += v * as_[d];
      dd += v * ad_[d];
    }
    esrc[bh * 1024 + n] = s;
    edst[bh * 1024 + n] = dd;
    tmax = fmaxf(tmax, dd);
  }
  red[t] = tmax;
  __syncthreads();
  for (int off = 128; off > 0; off >>= 1) {
    if (t < off) red[t] = fmaxf(red[t], red[t + off]);
    __syncthreads();
  }
  if (t == 0) maxd[bh] = red[0];
}

// ---------------------------------------------------------------------------
// Kernel 2: fused masked-softmax + alpha@Ht + ELU.
// One wave per (b,h, 16-row tile). Single pass over the mask A:
//   bound_n = leaky(e_src[n] + max_m e_dst[m])  >= true row max (monotone leaky)
//   p = A ? exp(leaky(e_src+e_dst) - bound) : 0   -> f16 A-operand
//   acc(16x64, f32) += p @ Ht via 4x V_WMMA_F32_16X16X32_F16 per 32 columns
//   out = ELU(acc / sum(p))
// ---------------------------------------------------------------------------
__global__ __launch_bounds__(128) void k_attn(const int* __restrict__ Am,
                                              const _Float16* __restrict__ HtT,
                                              const float* __restrict__ esrc,
                                              const float* __restrict__ edst,
                                              const float* __restrict__ maxd,
                                              float* __restrict__ out) {
  const int lane = threadIdx.x & 31, wave = threadIdx.x >> 5;
  const int bh   = blockIdx.x >> 4;               // 32 (b,h) pairs
  const int tile = (blockIdx.x & 15) * 4 + wave;  // 64 row tiles per (b,h)
  const int b = bh >> 2, h = bh & 3;
  const int row  = lane & 15;                     // A-operand row / B-operand col
  const int half = lane >> 4;

  const int n = tile * 16 + row;
  const float s = esrc[bh * 1024 + n];
  float Mx = s + maxd[bh];
  Mx = fmaxf(Mx, 0.f) + 0.2f * fminf(Mx, 0.f);    // leaky(upper bound)

  const int*      __restrict__ arow  = Am   + (size_t)b * NN * NN + (size_t)n * NN;
  const float*    __restrict__ drow  = edst + bh * 1024;
  const _Float16* __restrict__ hbase = HtT  + (size_t)bh * 64 * 1024;

  float l = 0.f;
  v8f acc0 = {}, acc1 = {}, acc2 = {}, acc3 = {};
  const int cb = half * 8;    // A-operand k-group column base
  const int kb = half * 8;    // B-operand k base

  for (int m0 = 0; m0 < 1024; m0 += 32) {
    __builtin_prefetch(arow + m0 + 32, 0, 1);     // global_prefetch_b8 next tile
    // mask + e_dst for this lane's 16 (row, k) slots
    const int4   mk0 = *(const int4*)  (arow + m0 + cb);
    const int4   mk1 = *(const int4*)  (arow + m0 + cb + 4);
    const int4   mk2 = *(const int4*)  (arow + m0 + cb + 16);
    const int4   mk3 = *(const int4*)  (arow + m0 + cb + 20);
    const float4 ed0 = *(const float4*)(drow + m0 + cb);
    const float4 ed1 = *(const float4*)(drow + m0 + cb + 4);
    const float4 ed2 = *(const float4*)(drow + m0 + cb + 16);
    const float4 ed3 = *(const float4*)(drow + m0 + cb + 20);

    auto pv = [&](int mk, float ed) -> _Float16 {
      float e = s + ed;
      e = fmaxf(e, 0.f) + 0.2f * fminf(e, 0.f);   // leaky_relu(0.2)
      const float pp = (mk != 0) ? __expf(e - Mx) : 0.f;
      l += pp;
      return (_Float16)pp;
    };

    v16h p;
    p[0]  = pv(mk0.x, ed0.x); p[1]  = pv(mk0.y, ed0.y);
    p[2]  = pv(mk0.z, ed0.z); p[3]  = pv(mk0.w, ed0.w);
    p[4]  = pv(mk1.x, ed1.x); p[5]  = pv(mk1.y, ed1.y);
    p[6]  = pv(mk1.z, ed1.z); p[7]  = pv(mk1.w, ed1.w);
    p[8]  = pv(mk2.x, ed2.x); p[9]  = pv(mk2.y, ed2.y);
    p[10] = pv(mk2.z, ed2.z); p[11] = pv(mk2.w, ed2.w);
    p[12] = pv(mk3.x, ed3.x); p[13] = pv(mk3.y, ed3.y);
    p[14] = pv(mk3.z, ed3.z); p[15] = pv(mk3.w, ed3.w);

    // B operand: lane = column d (row), k groups {kb..kb+7, kb+16..kb+23} along HtT's n axis
#define BTILE(ACC, T)                                                         \
    {                                                                         \
      const _Float16* hp = hbase + ((size_t)((T) * 16 + row)) * 1024 + m0 + kb;\
      union { int4 q[2]; v16h v; } u;                                         \
      u.q[0] = *(const int4*)hp;                                              \
      u.q[1] = *(const int4*)(hp + 16);                                       \
      ACC = __builtin_amdgcn_wmma_f32_16x16x32_f16(false, p, false, u.v,      \
                                                   (short)0, ACC, false, false);\
    }
    BTILE(acc0, 0)
    BTILE(acc1, 1)
    BTILE(acc2, 2)
    BTILE(acc3, 3)
#undef BTILE
  }

  // combine row-sums across half-waves (rows live in lanes r and r+16)
  l += __shfl_xor(l, 16, 32);
  const float inv = 1.0f / l;

#pragma unroll
  for (int i = 0; i < 8; ++i) {
    const float invr = __shfl(inv, i + half * 8, 32); // row of C/D VGPR i
    const int no = tile * 16 + i + half * 8;
    const size_t oidx = ((size_t)(b * 1024 + no)) * 256 + h * 64 + row;
    float o;
    o = acc0[i] * invr; out[oidx +  0] = o > 0.f ? o : __expf(o) - 1.f;
    o = acc1[i] * invr; out[oidx + 16] = o > 0.f ? o : __expf(o) - 1.f;
    o = acc2[i] * invr; out[oidx + 32] = o > 0.f ? o : __expf(o) - 1.f;
    o = acc3[i] * invr; out[oidx + 48] = o > 0.f ? o : __expf(o) - 1.f;
  }
}

// ---------------------------------------------------------------------------
extern "C" void kernel_launch(void* const* d_in, const int* in_sizes, int n_in,
                              void* d_out, int out_size, void* d_ws, size_t ws_size,
                              hipStream_t stream) {
  const float* Hm = (const float*)d_in[0];   // (8,1024,128) f32
  const int*   Am = (const int*)  d_in[1];   // (8,1024,1024) i32
  const float* Wm = (const float*)d_in[2];   // (128,256) f32
  const float* av = (const float*)d_in[3];   // (4,128) f32
  float* out = (float*)d_out;                // (8,1024,256) f32

  char* ws = (char*)d_ws;
  _Float16* HtT  = (_Float16*)ws;                          // 32*64*1024*2B = 4 MB
  float*    esrc = (float*)(ws + (size_t)4 * 1024 * 1024); // 128 KB
  float*    edst = esrc + 32 * 1024;                       // 128 KB
  float*    maxd = edst + 32 * 1024;                       // 128 B

  k_proj<<<2048, 128, 0, stream>>>(Hm, Wm, HtT);
  k_edge<<<32,   256, 0, stream>>>(HtT, av, esrc, edst, maxd);
  k_attn<<<512,  128, 0, stream>>>(Am, HtT, esrc, edst, maxd, out);
}
